// VectorQuantizer_15384572854332
// MI455X (gfx1250) — matrix operand order repaired and checked by reading, not compiled, not run on hardware
//
#include <hip/hip_runtime.h>
#include <hip/hip_bf16.h>

typedef __attribute__((ext_vector_type(16))) __bf16          v16bf;
typedef __attribute__((ext_vector_type(16))) unsigned short  v16u16;
typedef __attribute__((ext_vector_type(8)))  float           v8f;

#define VQ_B   32
#define VQ_C   512
#define VQ_HW  4096
#define VQ_K   512
#define VQ_N   (VQ_B * VQ_HW)          /* 131072 rows   */
#define VQ_M   ((long)VQ_N * VQ_C)     /* 67108864 elems */

/* workspace layout (byte offsets) */
#define WS_EBF16   0            /* 512*512*2  = 524288  */
#define WS_ENORM   524288       /* 512*4      = 2048    */
#define WS_IDX     526336       /* 131072*4   = 524288  */
#define WS_LOSS    1050624      /* f32 */
#define WS_PERP    1050628      /* f32 */

__device__ __forceinline__ unsigned short f2bf(float f) {
    unsigned u = __float_as_uint(f);
    unsigned r = u + 0x7FFFu + ((u >> 16) & 1u);   /* round-to-nearest-even */
    return (unsigned short)(r >> 16);
}

/* ---- 1) embed -> bf16, ||e||^2, zero accumulators ---------------------- */
__global__ void vq_prep(const float* __restrict__ embed,
                        unsigned short* __restrict__ ebf,
                        float* __restrict__ enorm,
                        float* loss_acc, float* perp_acc) {
    const int k = blockIdx.x, t = threadIdx.x;
    const float* row = embed + (size_t)k * VQ_C;
    float s = 0.f;
#pragma unroll
    for (int j = 0; j < 4; ++j) {
        float v = row[t * 4 + j];
        ebf[(size_t)k * VQ_C + t * 4 + j] = f2bf(v);
        s += v * v;
    }
    __shared__ float red[128];
    red[t] = s; __syncthreads();
    for (int o = 64; o; o >>= 1) { if (t < o) red[t] += red[t + o]; __syncthreads(); }
    if (t == 0) enorm[k] = red[0];
    if (k == 0 && t == 0) { *loss_acc = 0.f; *perp_acc = 0.f; }
}

/* ---- 2) argmin_k ( ||e_k||^2 - 2 x.e_k ) via bf16 WMMA ------------------
 * 32 rows per wave (two 16x512 A tiles register-resident, loaded straight
 * from global), B streamed from L2-resident bf16 codebook; each B fragment
 * feeds two WMMAs -> halves B bandwidth per FLOP vs M=16. Two independent
 * accumulator chains improve WMMA pipelining.                              */
__global__ void __launch_bounds__(64) vq_argmin(const float* __restrict__ x,
        const unsigned short* __restrict__ ebf,
        const float* __restrict__ enorm,
        int* __restrict__ idx) {
    const int lane = threadIdx.x & 31;
    const int wv   = threadIdx.x >> 5;
    const int rowbase = blockIdx.x * 64 + wv * 32;

    const int nlo = lane & 15, hf = lane >> 4;

    /* per-lane base pointers for the two M-tiles (row = rowbase + 16*t + nlo) */
    const float* xbase[2];
#pragma unroll
    for (int t = 0; t < 2; ++t) {
        const int n  = rowbase + 16 * t + nlo;
        const int b  = n >> 12, hw = n & 4095;
        xbase[t] = x + (size_t)(b * VQ_C) * VQ_HW + hw;
    }

    /* build register-resident A tiles: afrag[t][cc] holds
     * K = cc*32 + hf*16 + [0..7]  (low 8 halves)  and
     * K = cc*32 + hf*16 + 8 + [0..7] (high 8 halves), matching B's order   */
    v16bf afrag0[16], afrag1[16];
#pragma unroll
    for (int cc = 0; cc < 16; ++cc) {
        v16u16 t0, t1;
#pragma unroll
        for (int j = 0; j < 8; ++j) {
            const int c0 = cc * 32 + hf * 16 + j;
            const int c1 = cc * 32 + hf * 16 + 8 + j;
            t0[j]     = f2bf(xbase[0][(size_t)c0 * VQ_HW]);
            t0[8 + j] = f2bf(xbase[0][(size_t)c1 * VQ_HW]);
            t1[j]     = f2bf(xbase[1][(size_t)c0 * VQ_HW]);
            t1[8 + j] = f2bf(xbase[1][(size_t)c1 * VQ_HW]);
        }
        afrag0[cc] = __builtin_bit_cast(v16bf, t0);
        afrag1[cc] = __builtin_bit_cast(v16bf, t1);
    }

    float mv0[8], mv1[8]; int mi0[8], mi1[8];
#pragma unroll
    for (int r = 0; r < 8; ++r) {
        mv0[r] = 3.4e38f; mv1[r] = 3.4e38f; mi0[r] = 0; mi1[r] = 0;
    }

    for (int kt = 0; kt < VQ_K / 16; ++kt) {
        const int col = kt * 16 + nlo;
        const unsigned short* brow = ebf + (size_t)col * VQ_C + hf * 16;
        const float en = enorm[col];
        v8f acc0 = {}, acc1 = {};
#pragma unroll
        for (int cc = 0; cc < VQ_C / 32; ++cc) {
            /* B: lane holds N=col; K = hf*16 + [0..15], contiguous in row */
            v16bf bm = *(const v16bf*)(brow + cc * 32);
            acc0 = __builtin_amdgcn_wmma_f32_16x16x32_bf16(
                     false, afrag0[cc], false, bm, (short)0, acc0, false, false);
            acc1 = __builtin_amdgcn_wmma_f32_16x16x32_bf16(
                     false, afrag1[cc], false, bm, (short)0, acc1, false, false);
        }
#pragma unroll
        for (int r = 0; r < 8; ++r) {
            float s0 = en - 2.0f * acc0[r];
            float s1 = en - 2.0f * acc1[r];
            if (s0 < mv0[r]) { mv0[r] = s0; mi0[r] = col; }
            if (s1 < mv1[r]) { mv1[r] = s1; mi1[r] = col; }
        }
    }

    /* min-reduce over the 16 lanes of each half (N dimension) and write:
     * C/D layout: VGPR r -> row r (lanes 0-15) / row r+8 (lanes 16-31)    */
#pragma unroll
    for (int r = 0; r < 8; ++r) {
        float v0 = mv0[r], v1 = mv1[r]; int i0 = mi0[r], i1 = mi1[r];
#pragma unroll
        for (int off = 8; off; off >>= 1) {
            float ov0 = __shfl_xor(v0, off, 16);
            int   oi0 = __shfl_xor(i0, off, 16);
            float ov1 = __shfl_xor(v1, off, 16);
            int   oi1 = __shfl_xor(i1, off, 16);
            if (ov0 < v0 || (ov0 == v0 && oi0 < i0)) { v0 = ov0; i0 = oi0; }
            if (ov1 < v1 || (ov1 == v1 && oi1 < i1)) { v1 = ov1; i1 = oi1; }
        }
        if (nlo == 0) {
            idx[rowbase +      r + 8 * hf] = i0;
            idx[rowbase + 16 + r + 8 * hf] = i1;
        }
    }
}

/* ---- 3) gather codes into BCHW output + MSE loss ----------------------- */
__global__ void vq_quant(const float* __restrict__ x,
                         const float* __restrict__ embed,
                         const int* __restrict__ idx,
                         float* __restrict__ outq,
                         float* __restrict__ loss_acc) {
    const long i4 = ((long)blockIdx.x * blockDim.x + threadIdx.x) * 4;
    const int b  = (int)(i4 >> 21);          /* C*HW = 2^21 */
    const int c  = (int)((i4 >> 12) & 511);
    const int hw = (int)(i4 & 4095);
    float4 xv = *(const float4*)(x + i4);
    int4 cd = *(const int4*)(idx + b * VQ_HW + hw);
    float q0 = embed[(size_t)cd.x * VQ_C + c];
    float q1 = embed[(size_t)cd.y * VQ_C + c];
    float q2 = embed[(size_t)cd.z * VQ_C + c];
    float q3 = embed[(size_t)cd.w * VQ_C + c];
    outq[i4 + 0] = q0; outq[i4 + 1] = q1; outq[i4 + 2] = q2; outq[i4 + 3] = q3;
    float d0 = q0 - xv.x, d1 = q1 - xv.y, d2 = q2 - xv.z, d3 = q3 - xv.w;
    float s = d0*d0 + d1*d1 + d2*d2 + d3*d3;
    __shared__ float red[256];
    red[threadIdx.x] = s; __syncthreads();
    for (int o = 128; o; o >>= 1) { if (threadIdx.x < o) red[threadIdx.x] += red[threadIdx.x + o]; __syncthreads(); }
    if (threadIdx.x == 0) atomicAdd(loss_acc, red[0]);
}

/* ---- 4) per-batch unique-code count ----------------------------------- */
__global__ void vq_perp(const int* __restrict__ idx, float* __restrict__ perp_acc) {
    __shared__ int flags[VQ_K];
    const int b = blockIdx.x, t = threadIdx.x;
    flags[t] = 0; flags[t + 256] = 0;
    __syncthreads();
    for (int j = t; j < VQ_HW; j += 256) flags[idx[b * VQ_HW + j]] = 1;
    __syncthreads();
    __shared__ int red[256];
    red[t] = flags[t] + flags[t + 256]; __syncthreads();
    for (int o = 128; o; o >>= 1) { if (t < o) red[t] += red[t + o]; __syncthreads(); }
    if (t == 0) atomicAdd(perp_acc, (float)red[0]);
}

/* ---- 5) outputs -------------------------------------------------------- */
__global__ void vq_idxout(const int* __restrict__ idx, float* __restrict__ out) {
    const int t = blockIdx.x * blockDim.x + threadIdx.x;
    out[t] = (float)idx[t];
}
__global__ void vq_final(const float* loss_acc, const float* perp_acc,
                         float* __restrict__ out) {
    out[0] = 1.25f * (*loss_acc) / (float)VQ_M;      /* q_latent + 0.25*e_latent */
    out[1 + VQ_M] = (*perp_acc) / (float)VQ_B;
}

extern "C" void kernel_launch(void* const* d_in, const int* in_sizes, int n_in,
                              void* d_out, int out_size, void* d_ws, size_t ws_size,
                              hipStream_t stream) {
    const float* x     = (const float*)d_in[0];
    const float* embed = (const float*)d_in[1];
    float* out = (float*)d_out;
    char*  ws  = (char*)d_ws;
    unsigned short* ebf = (unsigned short*)(ws + WS_EBF16);
    float* enorm    = (float*)(ws + WS_ENORM);
    int*   idx      = (int*)(ws + WS_IDX);
    float* loss_acc = (float*)(ws + WS_LOSS);
    float* perp_acc = (float*)(ws + WS_PERP);

    vq_prep  <<<VQ_K, 128, 0, stream>>>(embed, ebf, enorm, loss_acc, perp_acc);
    vq_argmin<<<VQ_N / 64, 64, 0, stream>>>(x, ebf, enorm, idx);
    vq_quant <<<(int)(VQ_M / (256 * 4)), 256, 0, stream>>>(x, embed, idx, out + 1, loss_acc);
    vq_perp  <<<VQ_B, 256, 0, stream>>>(idx, perp_acc);
    vq_idxout<<<VQ_N / 256, 256, 0, stream>>>(idx, out + 2 + VQ_M);
    vq_final <<<1, 1, 0, stream>>>(loss_acc, perp_acc, out);
}
